// DeepRC_mod_33870112096285
// MI455X (gfx1250) — matrix operand order repaired and checked by reading, not compiled, not run on hardware
//
#include <hip/hip_runtime.h>
#include <stdint.h>

#define NSEQ   250000
#define NBAGS  100
#define TILES  (NSEQ / 16)   // 15625 wave-tiles, exact
#define NFRAG  18            // 6 conv1 + 8 conv2 + 2 att1 + 2 att2

typedef _Float16 v16h __attribute__((ext_vector_type(16)));
typedef float    v8f  __attribute__((ext_vector_type(8)));

union Frag16 { v16h v; _Float16 h[16]; uint2 u2[4]; uint4 u4[2]; };

__device__ __forceinline__ v8f wmma16(v16h a, v16h b, v8f c) {
  // D = A(16x32 f16) * B(32x16 f16) + C(16x16 f32)
  return __builtin_amdgcn_wmma_f32_16x16x32_f16(false, a, false, b, (short)0, c, false, false);
}

// Branch-free SELU: sc*max(v,0) + sc*al*(exp(min(v,0))-1); exact at v==0.
__device__ __forceinline__ float selu_f(float v) {
  const float sc = 1.0507009873554805f;
  const float scal = 1.0507009873554805f * 1.6732632423543772f;
  return sc * fmaxf(v, 0.f) + scal * (__expf(fminf(v, 0.f)) - 1.f);
}

__device__ __forceinline__ float sigmoid_f(float v) {
  return 1.f / (1.f + __expf(-v));
}

__device__ __forceinline__ v8f bcast8(float v) {
  v8f r;
#pragma unroll
  for (int i = 0; i < 8; ++i) r[i] = v;
  return r;
}

// ---------------------------------------------------------------------------
// Kernel 0: pack all WMMA B-fragments (32x16 f16, K x N) into workspace in
// per-lane layout [frag][lane][16 halves] so k_embed fetches each fragment
// with two global_load_b128.  B layout: lanes 0-15 hold K=0..7,16..23 of column
// N=lane(+16*nhalf), lanes 16-31 hold K=8..15,24..31.  W row-major [K][32].
// frag id: conv1 q*2+h (0..5, klimit 80), conv2 6+q*2+h (6..13, klimit 128),
//          att1 14+h, att2 16+h (klimit 32).
// ---------------------------------------------------------------------------
__global__ __launch_bounds__(288) void k_pack(
    const float* __restrict__ w1, const float* __restrict__ w2,
    const float* __restrict__ aw1, const float* __restrict__ aw2,
    _Float16* __restrict__ wp)
{
  const int idx = blockIdx.x * 288 + threadIdx.x;   // 0..575
  if (idx >= NFRAG * 32) return;
  const int f = idx >> 5, lane = idx & 31;
  const float* W; int kbase, klimit;
  if (f < 6)       { W = w1;  kbase = (f >> 1) * 32;       klimit = 80;  }
  else if (f < 14) { W = w2;  kbase = ((f - 6) >> 1) * 32; klimit = 128; }
  else if (f < 16) { W = aw1; kbase = 0;                   klimit = 32;  }
  else             { W = aw2; kbase = 0;                   klimit = 32;  }
  const int n    = (lane & 15) + 16 * (f & 1);
  const int koff = (lane < 16) ? 0 : 8;
  _Float16* dst = wp + (size_t)idx * 16;
#pragma unroll
  for (int j = 0; j < 16; ++j) {
    int k = kbase + koff + j + (j >= 8 ? 8 : 0);
    dst[j] = (_Float16)((k < klimit) ? W[k * 32 + n] : 0.f);
  }
}

// ---------------------------------------------------------------------------
// Kernel 1: fused conv1 + SELU + conv2 + SELU + length-max + attention MLP.
// One wave (32 threads) per block, 16 sequences per wave. 15625 blocks exact.
// Biases ride as the C operand of the first WMMA of each chain (free add).
// ---------------------------------------------------------------------------
__global__ __launch_bounds__(32) void k_embed(
    const float* __restrict__ x,
    const _Float16* __restrict__ wpack,
    const float* __restrict__ b1, const float* __restrict__ b2,
    const float* __restrict__ ab1, const float* __restrict__ ab2,
    const float* __restrict__ aw3, const float* __restrict__ ab3,
    float* __restrict__ emb, float* __restrict__ scores)
{
  __shared__ _Float16 xs[16 * 816];   // 16 seqs x (800 data + 16 zero pad)
  __shared__ _Float16 hb[16 * 128];   // conv1 ring: 16 seqs x 4 rows x 32 ch
  __shared__ _Float16 se[16 * 32];    // seq_embed / attention hidden (f16)
  __shared__ float    a2[16 * 32];    // attention layer-2 output (f32)

  const int lane  = threadIdx.x;
  const int tile  = blockIdx.x;
  const int s0    = tile * 16;
  const int m15   = lane & 15;
  const int mbase = (lane < 16) ? 0 : 8;   // C-layout row base
  const int koffA = (lane < 16) ? 0 : 8;   // A-layout K offset

  // fragment fetch: 2x b128 per fragment, fully coalesced
  auto ldfrag = [&](Frag16& fr, int f) {
    const uint4* p = (const uint4*)(wpack + ((size_t)f * 32 + lane) * 16);
    fr.u4[0] = p[0]; fr.u4[1] = p[1];
  };

  // ---- conv weights resident in VGPRs during the t loop ----
  Frag16 cw1[3][2], cw2[4][2];
#pragma unroll
  for (int q = 0; q < 3; ++q) { ldfrag(cw1[q][0], q*2); ldfrag(cw1[q][1], q*2+1); }
#pragma unroll
  for (int q = 0; q < 4; ++q) { ldfrag(cw2[q][0], 6+q*2); ldfrag(cw2[q][1], 6+q*2+1); }

  // per-channel biases broadcast into resident C-init vectors
  const v8f vb1[2] = { bcast8(b1[m15]), bcast8(b1[m15 + 16]) };
  const v8f vb2[2] = { bcast8(b2[m15]), bcast8(b2[m15 + 16]) };

  // ---- stage x tile -> LDS as f16 (zero-pad K tail: pad x 0-weight != NaN) ----
  for (int i = lane; i < 16 * 16; i += 32) {
    int s = i >> 4, j = i & 15;
    xs[s * 816 + 800 + j] = (_Float16)0.f;
  }
  for (int i = lane; i < 16 * 200; i += 32) {
    int s = i / 200, j = i % 200;
    const float4 v = *(const float4*)(x + (size_t)(s0 + s) * 800 + j * 4);
    union { _Float16 h[4]; uint2 u; } pk;
    pk.h[0] = (_Float16)v.x; pk.h[1] = (_Float16)v.y;
    pk.h[2] = (_Float16)v.z; pk.h[3] = (_Float16)v.w;
    *(uint2*)&xs[s * 816 + j * 4] = pk.u;
  }
  if (tile + 1 < TILES)  // hint next tile into cache (global_prefetch_b8)
    __builtin_prefetch(x + (size_t)(s0 + 16) * 800 + lane * 25, 0, 0);

  // conv1 position t -> SELU -> ring slot t&3   (bias pre-loaded into C)
  auto conv1_step = [&](int t) {
    v8f c0 = vb1[0], c1 = vb1[1];
#pragma unroll
    for (int q = 0; q < 3; ++q) {
      Frag16 a;
      const _Float16* p = &xs[m15 * 816 + t * 20 + q * 32 + koffA];
      a.u2[0] = *(const uint2*)(p);
      a.u2[1] = *(const uint2*)(p + 4);
      a.u2[2] = *(const uint2*)(p + 16);
      a.u2[3] = *(const uint2*)(p + 20);
      c0 = wmma16(a.v, cw1[q][0].v, c0);
      c1 = wmma16(a.v, cw1[q][1].v, c1);
    }
    const int slot = t & 3;
#pragma unroll
    for (int i = 0; i < 8; ++i) {
      hb[(mbase + i) * 128 + slot * 32 + m15]      = (_Float16)selu_f(c0[i]);
      hb[(mbase + i) * 128 + slot * 32 + m15 + 16] = (_Float16)selu_f(c1[i]);
    }
  };

  v8f mx0, mx1;
#pragma unroll
  for (int i = 0; i < 8; ++i) { mx0[i] = -3.4e38f; mx1[i] = -3.4e38f; }

  // conv2 position tp -> SELU -> running max (C layout is position-invariant)
  auto conv2_step = [&](int tp) {
    v8f d0 = vb2[0], d1 = vb2[1];
#pragma unroll
    for (int q = 0; q < 4; ++q) {
      Frag16 a;
      const _Float16* p = &hb[m15 * 128 + ((tp + q) & 3) * 32 + koffA];
      a.u4[0] = *(const uint4*)(p);
      a.u4[1] = *(const uint4*)(p + 16);
      d0 = wmma16(a.v, cw2[q][0].v, d0);
      d1 = wmma16(a.v, cw2[q][1].v, d1);
    }
#pragma unroll
    for (int i = 0; i < 8; ++i) {
      mx0[i] = fmaxf(mx0[i], selu_f(d0[i]));
      mx1[i] = fmaxf(mx1[i], selu_f(d1[i]));
    }
  };

  // prologue: fill ring rows 0..2, then steady state (no branch in body)
#pragma unroll
  for (int t = 0; t < 3; ++t) conv1_step(t);
#pragma unroll 2
  for (int tp = 0; tp < 34; ++tp) {
    conv1_step(tp + 3);
    conv2_step(tp);
  }

  // ---- seq_embed -> global + LDS (f16) for attention ----
#pragma unroll
  for (int i = 0; i < 8; ++i) {
    int m = mbase + i;
    se[m * 32 + m15]      = (_Float16)mx0[i];
    se[m * 32 + m15 + 16] = (_Float16)mx1[i];
    emb[(size_t)(s0 + m) * 32 + m15]      = mx0[i];
    emb[(size_t)(s0 + m) * 32 + m15 + 16] = mx1[i];
  }

  // attention fragments/biases loaded late to keep conv-loop VGPR pressure low
  Frag16 fa1[2], fa2[2];
  ldfrag(fa1[0], 14); ldfrag(fa1[1], 15);
  ldfrag(fa2[0], 16); ldfrag(fa2[1], 17);
  const v8f vba1[2] = { bcast8(ab1[m15]), bcast8(ab1[m15 + 16]) };
  const v8f vba2[2] = { bcast8(ab2[m15]), bcast8(ab2[m15 + 16]) };

  // ---- attention layer 1: 16x32 @ 32x32, SELU ----
  {
    Frag16 a;
    const _Float16* p = &se[m15 * 32 + koffA];
    a.u4[0] = *(const uint4*)(p);
    a.u4[1] = *(const uint4*)(p + 16);
    v8f e0 = wmma16(a.v, fa1[0].v, vba1[0]);
    v8f e1 = wmma16(a.v, fa1[1].v, vba1[1]);
#pragma unroll
    for (int i = 0; i < 8; ++i) {
      int m = mbase + i;
      se[m * 32 + m15]      = (_Float16)selu_f(e0[i]);
      se[m * 32 + m15 + 16] = (_Float16)selu_f(e1[i]);
    }
  }
  // ---- attention layer 2: 16x32 @ 32x32, SELU (f32 out for the final dot) ----
  {
    Frag16 a;
    const _Float16* p = &se[m15 * 32 + koffA];
    a.u4[0] = *(const uint4*)(p);
    a.u4[1] = *(const uint4*)(p + 16);
    v8f e0 = wmma16(a.v, fa2[0].v, vba2[0]);
    v8f e1 = wmma16(a.v, fa2[1].v, vba2[1]);
#pragma unroll
    for (int i = 0; i < 8; ++i) {
      int m = mbase + i;
      a2[m * 32 + m15]      = selu_f(e0[i]);
      a2[m * 32 + m15 + 16] = selu_f(e1[i]);
    }
  }
  // ---- attention layer 3: 32 -> 1 ----
  if (lane < 16) {
    float s = ab3[0];
#pragma unroll
    for (int c = 0; c < 32; ++c) s += a2[lane * 32 + c] * aw3[c];
    scores[s0 + lane] = s;
  }
}

// ---------------------------------------------------------------------------
// Kernel 2: per-bag segment softmax + weighted-sum pooling. 1 block per bag.
// ---------------------------------------------------------------------------
__global__ __launch_bounds__(256) void k_pool(
    const float* __restrict__ scores, const float* __restrict__ emb,
    const int* __restrict__ nper, float* __restrict__ pooled)
{
  const int bag = blockIdx.x, tid = threadIdx.x;
  __shared__ float red[256];
  __shared__ float pool[32];
  __shared__ int   sb[2];

  if (tid == 0) {
    int b = 0;
    for (int i = 0; i < bag; ++i) b += nper[i];
    sb[0] = b; sb[1] = nper[bag];
  }
  __syncthreads();
  const int base = sb[0], n = sb[1];

  // segment max
  float m = -3.4e38f;
  for (int i = tid; i < n; i += 256) m = fmaxf(m, scores[base + i]);
  red[tid] = m; __syncthreads();
  for (int s = 128; s > 0; s >>= 1) {
    if (tid < s) red[tid] = fmaxf(red[tid], red[tid + s]);
    __syncthreads();
  }
  const float mx = red[0]; __syncthreads();

  // segment sum of exp
  float sum = 0.f;
  for (int i = tid; i < n; i += 256) sum += __expf(scores[base + i] - mx);
  red[tid] = sum; __syncthreads();
  for (int s = 128; s > 0; s >>= 1) {
    if (tid < s) red[tid] += red[tid + s];
    __syncthreads();
  }
  const float denom = red[0]; __syncthreads();

  if (tid < 32) pool[tid] = 0.f;
  __syncthreads();

  float4 acc[8] = {};
  for (int i = tid; i < n; i += 256) {
    const float w = __expf(scores[base + i] - mx) / denom;
    const float4* e4 = (const float4*)&emb[(size_t)(base + i) * 32];
#pragma unroll
    for (int c = 0; c < 8; ++c) {
      float4 e = e4[c];
      acc[c].x += w * e.x; acc[c].y += w * e.y;
      acc[c].z += w * e.z; acc[c].w += w * e.w;
    }
  }
#pragma unroll
  for (int c = 0; c < 8; ++c) {
    atomicAdd(&pool[4 * c + 0], acc[c].x);
    atomicAdd(&pool[4 * c + 1], acc[c].y);
    atomicAdd(&pool[4 * c + 2], acc[c].z);
    atomicAdd(&pool[4 * c + 3], acc[c].w);
  }
  __syncthreads();
  if (tid < 32) pooled[bag * 32 + tid] = pool[tid];
}

// ---------------------------------------------------------------------------
// Kernel 3: output heads (100x32 -> 100x21 and 100x40), sigmoid. Tiny.
// ---------------------------------------------------------------------------
__global__ __launch_bounds__(64) void k_out(
    const float* __restrict__ pooled,
    const float* __restrict__ Aw1, const float* __restrict__ Ab1,
    const float* __restrict__ Aw2, const float* __restrict__ Ab2,
    const float* __restrict__ Bw1, const float* __restrict__ Bb1,
    const float* __restrict__ Bw2, const float* __restrict__ Bb2,
    float* __restrict__ out)
{
  const int bag = blockIdx.x, tid = threadIdx.x;
  __shared__ float p[32], hA[32], hB[32];
  if (tid < 32) p[tid] = pooled[bag * 32 + tid];
  __syncthreads();
  if (tid < 32) {
    float sa = Ab1[tid], sb = Bb1[tid];
#pragma unroll
    for (int k = 0; k < 32; ++k) {
      sa += p[k] * Aw1[k * 32 + tid];
      sb += p[k] * Bw1[k * 32 + tid];
    }
    hA[tid] = selu_f(sa); hB[tid] = selu_f(sb);
  }
  __syncthreads();
  if (tid < 21) {
    float s = Ab2[tid];
#pragma unroll
    for (int k = 0; k < 32; ++k) s += hA[k] * Aw2[k * 21 + tid];
    out[bag * 21 + tid] = sigmoid_f(s);
  }
  if (tid < 40) {
    float s = Bb2[tid];
#pragma unroll
    for (int k = 0; k < 32; ++k) s += hB[k] * Bw2[k * 40 + tid];
    out[NBAGS * 21 + bag * 40 + tid] = sigmoid_f(s);
  }
}

// ---------------------------------------------------------------------------
extern "C" void kernel_launch(void* const* d_in, const int* in_sizes, int n_in,
                              void* d_out, int out_size, void* d_ws, size_t ws_size,
                              hipStream_t stream) {
  const float* x    = (const float*)d_in[0];
  const int*   nper = (const int*)  d_in[1];
  const float* w1   = (const float*)d_in[2];
  const float* b1   = (const float*)d_in[3];
  const float* w2   = (const float*)d_in[4];
  const float* b2   = (const float*)d_in[5];
  const float* aw1  = (const float*)d_in[6];
  const float* ab1  = (const float*)d_in[7];
  const float* aw2  = (const float*)d_in[8];
  const float* ab2  = (const float*)d_in[9];
  const float* aw3  = (const float*)d_in[10];
  const float* ab3  = (const float*)d_in[11];
  const float* Aw1  = (const float*)d_in[12];
  const float* Ab1  = (const float*)d_in[13];
  const float* Aw2  = (const float*)d_in[14];
  const float* Ab2  = (const float*)d_in[15];
  const float* Bw1  = (const float*)d_in[16];
  const float* Bb1  = (const float*)d_in[17];
  const float* Bw2  = (const float*)d_in[18];
  const float* Bb2  = (const float*)d_in[19];

  char* ws = (char*)d_ws;
  float*    emb    = (float*)(ws);                  // 250000*32*4 = 32,000,000 B
  float*    scores = (float*)(ws + 32000000);       //  250000*4   =  1,000,000 B
  float*    pooled = (float*)(ws + 33000000);       //  100*32*4   =     12,800 B
  _Float16* wpack  = (_Float16*)(ws + 33020000);    //  18*32*16*2 =     18,432 B

  k_pack<<<2, 288, 0, stream>>>(w1, w2, aw1, aw2, wpack);
  k_embed<<<TILES, 32, 0, stream>>>(x, wpack, b1, b2, ab1, ab2, aw3, ab3,
                                    emb, scores);
  k_pool<<<NBAGS, 256, 0, stream>>>(scores, emb, nper, pooled);
  k_out<<<NBAGS, 64, 0, stream>>>(pooled, Aw1, Ab1, Aw2, Ab2,
                                  Bw1, Bb1, Bw2, Bb2, (float*)d_out);
}
